// TensorProductScoreModel_81406810128548
// MI455X (gfx1250) — compile-verified, matrix operand order checked by reading, not run on hardware
//
#include <hip/hip_runtime.h>
#include <math.h>

typedef __attribute__((ext_vector_type(16))) _Float16 v16h;
typedef __attribute__((ext_vector_type(8)))  _Float16 v8h;
typedef __attribute__((ext_vector_type(4)))  _Float16 v4h;
typedef __attribute__((ext_vector_type(8)))  float    v8f;

#define NS 16
#define NV 4
#define C_OUT 40
#define TDIM 8
#define EPSV 1e-5f

// ---------------- K0: zero scratch ----------------
__global__ void k_zero(float* p, long long n) {
    long long i = (long long)blockIdx.x * blockDim.x + threadIdx.x;
    if (i < n) p[i] = 0.0f;
}

// ---------------- K1: spectral conv (2 modes of rfft over T=8) ----------------
__global__ void k_spectral(const float* __restrict__ x,
                           const float* __restrict__ wr,
                           const float* __restrict__ wi,
                           float* __restrict__ xs, int N) {
    int id = blockIdx.x * blockDim.x + threadIdx.x;
    if (id >= N * NS) return;
    int n = id / NS, o = id % NS;

    float re0 = 0.f, re1 = 0.f, im1 = 0.f;
    const float ct[TDIM] = {1.f, 0.70710678f, 0.f, -0.70710678f, -1.f, -0.70710678f, 0.f, 0.70710678f};
    const float st[TDIM] = {0.f, 0.70710678f, 1.f, 0.70710678f, 0.f, -0.70710678f, -1.f, -0.70710678f};
    #pragma unroll
    for (int i = 0; i < NS; ++i) {
        float s = 0.f, a = 0.f, b = 0.f;
        #pragma unroll
        for (int t = 0; t < TDIM; ++t) {
            float xv = x[(long long)t * N * 28 + (long long)n * 28 + i];
            s += xv; a += xv * ct[t]; b += xv * st[t];
        }
        float imx = -b; // xf1 = a - i*b
        float wr0 = wr[(i * NS + o) * 2 + 0];
        float wr1 = wr[(i * NS + o) * 2 + 1], wi1 = wi[(i * NS + o) * 2 + 1];
        re0 += s * wr0;
        re1 += a * wr1 - imx * wi1;
        im1 += a * wi1 + imx * wr1;
    }
    #pragma unroll
    for (int t = 0; t < TDIM; ++t) {
        float v = (re0 + 2.f * (re1 * ct[t] - im1 * st[t])) * (1.0f / 8.0f);
        xs[(long long)t * N * NS + (long long)n * NS + o] = v;
    }
}

// ---------------- A-fragment: two contiguous b128 LDS loads ----------------
__device__ inline v16h ld_afrag(const _Float16* base, int stride, int row, int kbase, int hi) {
    const _Float16* p = base + row * stride + kbase + hi * 8;
    v8h lo = *(const v8h*)(p);
    v8h hh = *(const v8h*)(p + 16);
    return __builtin_shufflevector(lo, hh, 0, 1, 2, 3, 4, 5, 6, 7,
                                           8, 9, 10, 11, 12, 13, 14, 15);
}

// ---------------- K2: fused MLP (WMMA) + tensor product + scatter ----------------
__global__ __launch_bounds__(128, 1)
void k_mlp_tp(const float* __restrict__ ea,   // (T,E,48)
              const float* __restrict__ w1, const float* __restrict__ b1,
              const float* __restrict__ w2, const float* __restrict__ b2,
              const float* __restrict__ x,    // (T,N,28)
              const int*   __restrict__ eidx, // (2,E)
              const float* __restrict__ esh,  // (T,E,4)
              float* __restrict__ seg, float* __restrict__ cnt,
              int N, int E, int TE) {
    // weights pre-swizzled into B-fragment-major layout: [tile][lane][16 halves]
    __shared__ _Float16 sW1f[ 6 * 32 * 16];
    __shared__ _Float16 sW2f[52 * 32 * 16];
    __shared__ _Float16 sA [64 * 64];
    __shared__ _Float16 sH [64 * 64];
    __shared__ _Float16 sEW[64 * 416];
    __shared__ float    sB1[48];
    __shared__ float    sB2[416];

    const int tid  = threadIdx.x;
    const int lane = tid & 31;
    const int wave = tid >> 5;
    const int rowBase = blockIdx.x * 64;

    if (tid == 0) { __builtin_prefetch(w2, 0, 1); __builtin_prefetch(ea + (long long)rowBase * 48, 0, 1); }

    // ---- stage weights into fragment-major layout (f32->f16, zero-pad K to 64) ----
    for (int i = tid; i < 6 * 512; i += 128) {
        int tl = i >> 9, l = (i >> 4) & 31, h = i & 15;
        int nt = tl >> 1, ks = tl & 1;
        int n  = nt * 16 + (l & 15);
        int vv = h >> 1;
        int k  = ks * 32 + ((vv >= 4) ? 16 : 0) + (l >> 4) * 8 + (vv & 3) * 2 + (h & 1);
        sW1f[i] = (k < 48) ? (_Float16)w1[k * 48 + n] : (_Float16)0.f;
    }
    for (int i = tid; i < 52 * 512; i += 128) {
        int tl = i >> 9, l = (i >> 4) & 31, h = i & 15;
        int nt = tl >> 1, ks = tl & 1;
        int n  = nt * 16 + (l & 15);
        int vv = h >> 1;
        int k  = ks * 32 + ((vv >= 4) ? 16 : 0) + (l >> 4) * 8 + (vv & 3) * 2 + (h & 1);
        sW2f[i] = (k < 48) ? (_Float16)w2[k * 416 + n] : (_Float16)0.f;
    }
    for (int i = tid; i < 64 * 64; i += 128) {
        int r = i >> 6, k = i & 63;
        int g = rowBase + r;
        sA[i] = (k < 48 && g < TE) ? (_Float16)ea[(long long)g * 48 + k] : (_Float16)0.f;
        sH[i] = (_Float16)0.f;
    }
    for (int i = tid; i < 48;  i += 128) sB1[i] = b1[i];
    for (int i = tid; i < 416; i += 128) sB2[i] = b2[i];
    __syncthreads();

    const int wrow0 = wave * 16;                  // this wave's 16 rows within block
    const int m  = lane & 15;                     // A-frag row for this lane
    const int hi = lane >> 4;
    const int cN = lane & 15, cHi = hi * 8;       // C/D layout

    // ---- MLP1: [16x64] @ [64x48] -> relu -> sH (A-frags hoisted) ----
    {
        v16h a0 = ld_afrag(sA, 64, wrow0 + m, 0,  hi);
        v16h a1 = ld_afrag(sA, 64, wrow0 + m, 32, hi);
        for (int nt = 0; nt < 3; ++nt) {
            v8f acc = {};
            v16h b0 = *(const v16h*)&sW1f[((nt * 2 + 0) * 32 + lane) * 16];
            v16h b1v = *(const v16h*)&sW1f[((nt * 2 + 1) * 32 + lane) * 16];
            acc = __builtin_amdgcn_wmma_f32_16x16x32_f16(false, a0, false, b0,
                                                         (short)0, acc, false, false);
            acc = __builtin_amdgcn_wmma_f32_16x16x32_f16(false, a1, false, b1v,
                                                         (short)0, acc, false, false);
            int n = nt * 16 + cN;
            #pragma unroll
            for (int r = 0; r < 8; ++r) {
                float h = acc[r] + sB1[n];
                h = h > 0.f ? h : 0.f;
                sH[(wrow0 + r + cHi) * 64 + n] = (_Float16)h;
            }
        }
    }
    __syncthreads();

    // ---- MLP2: [16x64] @ [64x416] -> sEW (A-frags hoisted) ----
    {
        v16h a0 = ld_afrag(sH, 64, wrow0 + m, 0,  hi);
        v16h a1 = ld_afrag(sH, 64, wrow0 + m, 32, hi);
        for (int nt = 0; nt < 26; ++nt) {
            v8f acc = {};
            v16h b0 = *(const v16h*)&sW2f[((nt * 2 + 0) * 32 + lane) * 16];
            v16h b1v = *(const v16h*)&sW2f[((nt * 2 + 1) * 32 + lane) * 16];
            acc = __builtin_amdgcn_wmma_f32_16x16x32_f16(false, a0, false, b0,
                                                         (short)0, acc, false, false);
            acc = __builtin_amdgcn_wmma_f32_16x16x32_f16(false, a1, false, b1v,
                                                         (short)0, acc, false, false);
            int n = nt * 16 + cN;
            #pragma unroll
            for (int r = 0; r < 8; ++r)
                sEW[(wrow0 + r + cHi) * 416 + n] = (_Float16)(acc[r] + sB2[n]);
        }
    }
    __syncthreads();

    // ---- per-row tensor product + scatter (vectorized LDS reads) ----
    if (tid < 64) {
        const int r = tid, g = rowBase + r;
        if (g < TE) {
            const int t = g / E, e = g - t * E;
            const int src = eidx[e], dst = eidx[E + e];
            const float inv3 = 0.57735027f, inv6 = 0.40824829f;
            const float n0 = 0.22360680f, n1o = 0.22360680f, n1e = 0.5f;
            const _Float16* ew  = &sEW[r * 416];
            const v16h*     ew16 = (const v16h*)ew;   // 16-half chunks (32B aligned)
            const v4h*      ew4  = (const v4h*) ew;   // 4-half chunks (8B aligned)
            const float* xr = &x[((long long)t * N + src) * 28];

            float s[NS], v[NV][3], sh1[3];
            #pragma unroll
            for (int i = 0; i < NS; ++i) s[i] = xr[i];
            #pragma unroll
            for (int i = 0; i < NV; ++i)
                for (int d = 0; d < 3; ++d) v[i][d] = xr[NS + i * 3 + d];
            const float sh0 = esh[(long long)g * 4 + 0];
            #pragma unroll
            for (int d = 0; d < 3; ++d) sh1[d] = esh[(long long)g * 4 + 1 + d];

            float vdots[NV], cr[NV][3];
            #pragma unroll
            for (int i = 0; i < NV; ++i) {
                vdots[i] = (v[i][0] * sh1[0] + v[i][1] * sh1[1] + v[i][2] * sh1[2]) * inv3;
                cr[i][0] = (v[i][1] * sh1[2] - v[i][2] * sh1[1]) * inv6;
                cr[i][1] = (v[i][2] * sh1[0] - v[i][0] * sh1[2]) * inv6;
                cr[i][2] = (v[i][0] * sh1[1] - v[i][1] * sh1[0]) * inv6;
            }

            // out0: w00 at chunks 0..15, w110 at chunks 21..24
            float out0[NS];
            #pragma unroll
            for (int o = 0; o < NS; ++o) out0[o] = 0.f;
            #pragma unroll 4
            for (int i = 0; i < NS; ++i) {
                v16h w = ew16[i];
                float si = s[i] * sh0;
                #pragma unroll
                for (int o = 0; o < NS; ++o) out0[o] += si * (float)w[o];
            }
            #pragma unroll
            for (int i = 0; i < NV; ++i) {
                v16h w = ew16[21 + i];
                #pragma unroll
                for (int o = 0; o < NS; ++o) out0[o] += vdots[i] * (float)w[o];
            }

            // q[j] from w01 (v4h chunks 64..79)
            float q[NV] = {0.f, 0.f, 0.f, 0.f};
            #pragma unroll 4
            for (int i = 0; i < NS; ++i) {
                v4h w = ew4[64 + i];
                #pragma unroll
                for (int j = 0; j < NV; ++j) q[j] += s[i] * (float)w[j];
            }

            // p3 from w10 (chunks 80..83), pe from w111 (chunks 100..103)
            float p3[NV][3], pe[NV][3];
            #pragma unroll
            for (int j = 0; j < NV; ++j)
                for (int d = 0; d < 3; ++d) { p3[j][d] = 0.f; pe[j][d] = 0.f; }
            #pragma unroll
            for (int i = 0; i < NV; ++i) {
                v4h w10r = ew4[80 + i];
                v4h w11r = ew4[100 + i];
                #pragma unroll
                for (int j = 0; j < NV; ++j) {
                    float w10ij = (float)w10r[j], w11ij = (float)w11r[j];
                    #pragma unroll
                    for (int d = 0; d < 3; ++d) {
                        p3[j][d] += v[i][d] * sh0 * w10ij;
                        pe[j][d] += cr[i][d] * w11ij;
                    }
                }
            }

            float msg[C_OUT];
            #pragma unroll
            for (int o = 0; o < NS; ++o) msg[o] = out0[o] * n0;
            #pragma unroll
            for (int j = 0; j < NV; ++j)
                for (int d = 0; d < 3; ++d) {
                    msg[NS + j * 3 + d]      = (q[j] * sh1[d] * inv3 + p3[j][d] * inv3) * n1o;
                    msg[NS + 12 + j * 3 + d] = pe[j][d] * n1e;
                }

            float* segrow = &seg[((long long)t * N + dst) * C_OUT];
            #pragma unroll 8
            for (int c = 0; c < C_OUT; ++c) atomicAdd(&segrow[c], msg[c]);
            if (t == 0) atomicAdd(&cnt[dst], 1.0f);
        }
    }
}

// ---------------- K3: mean + residual + BN statistics ----------------
__global__ void k_bnstats(float* __restrict__ seg, const float* __restrict__ cnt,
                          const float* __restrict__ xs,
                          float* __restrict__ sum, float* __restrict__ sumsq,
                          int N, long long total) {
    __shared__ float ls[C_OUT], lq[C_OUT];
    if (threadIdx.x < C_OUT) { ls[threadIdx.x] = 0.f; lq[threadIdx.x] = 0.f; }
    __syncthreads();
    long long idx = (long long)blockIdx.x * blockDim.x + threadIdx.x;
    if (idx < total) {
        long long nc = (long long)N * C_OUT;
        int t = (int)(idx / nc);
        long long rem = idx - (long long)t * nc;
        int n = (int)(rem / C_OUT), c = (int)(rem - (long long)n * C_OUT);
        float cv = cnt[n];
        float y = seg[idx] / (cv > 1.f ? cv : 1.f);
        if (c < NS) y += xs[((long long)t * N + n) * NS + c];
        seg[idx] = y;
        int k = (c * 8 + t) % C_OUT;
        atomicAdd(&ls[k], y);
        atomicAdd(&lq[k], y * y);
    }
    __syncthreads();
    if (threadIdx.x < C_OUT) {
        atomicAdd(&sum[threadIdx.x],   ls[threadIdx.x]);
        atomicAdd(&sumsq[threadIdx.x], lq[threadIdx.x]);
    }
}

// ---------------- K4: batch-norm finalize ----------------
__global__ void k_bnapply(const float* __restrict__ seg,
                          const float* __restrict__ sum, const float* __restrict__ sumsq,
                          const float* __restrict__ gamma, const float* __restrict__ beta,
                          float* __restrict__ out, int N, long long total) {
    long long idx = (long long)blockIdx.x * blockDim.x + threadIdx.x;
    if (idx >= total) return;
    long long nc = (long long)N * C_OUT;
    int t = (int)(idx / nc);
    long long rem = idx - (long long)t * nc;
    int c = (int)(rem % C_OUT);
    int k = (c * 8 + t) % C_OUT;
    float inv_cnt = 1.0f / (8.0f * (float)N);
    float mu  = sum[k] * inv_cnt;
    float var = sumsq[k] * inv_cnt - mu * mu;
    out[idx] = (seg[idx] - mu) * rsqrtf(var + EPSV) * gamma[k] + beta[k];
}

// ---------------- launcher ----------------
extern "C" void kernel_launch(void* const* d_in, const int* in_sizes, int n_in,
                              void* d_out, int out_size, void* d_ws, size_t ws_size,
                              hipStream_t stream) {
    const float* x    = (const float*)d_in[0];
    const int*   eidx = (const int*)  d_in[1];
    const float* ea   = (const float*)d_in[2];
    const float* esh  = (const float*)d_in[3];
    const float* w1   = (const float*)d_in[4];
    const float* b1   = (const float*)d_in[5];
    const float* w2   = (const float*)d_in[6];
    const float* b2   = (const float*)d_in[7];
    const float* wr   = (const float*)d_in[8];
    const float* wi   = (const float*)d_in[9];
    const float* gam  = (const float*)d_in[10];
    const float* bet  = (const float*)d_in[11];

    const int N  = in_sizes[0] / (TDIM * 28);
    const int E  = in_sizes[1] / 2;
    const int TE = TDIM * E;

    float* ws    = (float*)d_ws;
    float* seg   = ws;                                 // T*N*40
    float* cnt   = seg + (long long)TDIM * N * C_OUT;  // N
    float* xs    = cnt + N;                            // T*N*16
    float* sum   = xs  + (long long)TDIM * N * NS;     // 40
    float* sumsq = sum + C_OUT;                        // 40

    k_zero<<<(int)(((long long)TDIM * N * C_OUT + N + 255) / 256), 256, 0, stream>>>(
        seg, (long long)TDIM * N * C_OUT + N);
    k_zero<<<1, 2 * C_OUT, 0, stream>>>(sum, 2 * C_OUT);
    (void)ws_size; (void)n_in;

    k_spectral<<<(N * NS + 255) / 256, 256, 0, stream>>>(x, wr, wi, xs, N);

    k_mlp_tp<<<(TE + 63) / 64, 128, 0, stream>>>(ea, w1, b1, w2, b2, x, eidx, esh,
                                                 seg, cnt, N, E, TE);

    long long total = (long long)TDIM * N * C_OUT;
    k_bnstats<<<(int)((total + 255) / 256), 256, 0, stream>>>(seg, cnt, xs, sum, sumsq, N, total);
    k_bnapply<<<(int)((total + 255) / 256), 256, 0, stream>>>(seg, sum, sumsq, gam, bet,
                                                              (float*)d_out, N, total);
    (void)out_size;
}
